// LocalGatedPropagation_90108413870747
// MI455X (gfx1250) — matrix operand order repaired and checked by reading, not compiled, not run on hardware
//
#include <hip/hip_runtime.h>

#define N_    4
#define NH    2
#define DA    64
#define DV    512
#define HH    96
#define WW    96
#define HW    9216
#define WSQ   225     // 15*15 window
#define WSP   240     // padded window: slot = dy*16 + (dx+7), slot15 == 0
#define HID_  256

typedef __attribute__((ext_vector_type(16))) __bf16 v16bf;
typedef __attribute__((ext_vector_type(8)))  float  v8f;

__device__ __forceinline__ unsigned short f2bfu(float f) {
    union { float f; unsigned u; } a; a.f = f;
    unsigned r = a.u + 0x7FFFu + ((a.u >> 16) & 1u);   // round-to-nearest-even
    return (unsigned short)(r >> 16);
}
__device__ __forceinline__ __bf16 u2bf(unsigned short h) {
    return __builtin_bit_cast(__bf16, h);
}
__device__ __forceinline__ v8f wmma_bf(v16bf a, v16bf b, v8f c) {
    return __builtin_amdgcn_wmma_f32_16x16x32_bf16(false, a, false, b, (short)0, c, false, false);
}

// ---------------------------------------------------------------------------
// Kernel 1: scores = q/T (.) k  (local corr, WMMA)  + rel (grouped 1x1, WMMA)
//           + bias + validity mask + softmax  -> attn bf16 [n,g,hw,240]
// one wave per (n, g, y, x-tile of 16)
// ---------------------------------------------------------------------------
__global__ __launch_bounds__(32) void k_scores(const float* __restrict__ q,
                                               const float* __restrict__ k,
                                               const float* __restrict__ w_rel,
                                               const float* __restrict__ b_rel,
                                               unsigned short* __restrict__ attn)
{
    __shared__ float sc[16 * WSP];

    int b = blockIdx.x;
    const int xt = b % 6;  b /= 6;
    const int y  = b % 96; b /= 96;
    const int g  = b % 2;
    const int n  = b / 2;

    const int lane = threadIdx.x;
    const int l15  = lane & 15;
    const int kbA  = (lane >> 4) << 3;   // unused; A fragment also uses kbB (0 or 16)
    const int kbB  = (lane >> 4) << 4;   // B-fragment K base: 0 or 16
    const int x0   = xt * 16;

    for (int i = lane; i < 16 * WSP; i += 32) sc[i] = 0.f;
    __syncthreads();

    // ---- A = q/T, rows M=pos, K=channel (2 chunks of 32) ----
    v16bf a0, a1;
    {
        const float* qb = q + ((size_t)((n * NH + g) * DA)) * HW + y * WW + (x0 + l15);
        #pragma unroll
        for (int e = 0; e < 16; ++e) {
            int c = kbB + e;
            a0[e] = u2bf(f2bfu(qb[(size_t)c * HW] * 0.125f));
            a1[e] = u2bf(f2bfu(qb[(size_t)(c + 32) * HW] * 0.125f));
        }
    }

    // ---- relative embedding GEMM: rel[pos, widx] = (q/T)[pos,c] * (T*w_rel)[widx,c] ----
    #pragma unroll 1
    for (int wt = 0; wt < 15; ++wt) {
        const int widx = wt * 16 + l15;
        v16bf b0, b1;
        if (widx < WSQ) {
            const float* wb = w_rel + ((size_t)(g * WSQ + widx)) * DA + kbB;
            #pragma unroll
            for (int e = 0; e < 16; ++e) {
                b0[e] = u2bf(f2bfu(wb[e] * 8.0f));
                b1[e] = u2bf(f2bfu(wb[e + 32] * 8.0f));
            }
        } else {
            #pragma unroll
            for (int e = 0; e < 16; ++e) { b0[e] = u2bf(0); b1[e] = u2bf(0); }
        }
        v8f acc = {};
        acc = wmma_bf(a0, b0, acc);
        acc = wmma_bf(a1, b1, acc);
        if (widx < WSQ) {
            const int slot = (widx / 15) * 16 + (widx % 15);
            const int mh = (lane >> 4) * 8;
            #pragma unroll
            for (int r = 0; r < 8; ++r) sc[(r + mh) * WSP + slot] = acc[r];
        }
    }
    __syncthreads();

    // ---- local correlation: out[pos, kx] = sum_c qs[pos,c] * k[c, ky, kx] ----
    #pragma unroll 1
    for (int dy = 0; dy < 15; ++dy) {
        const int ky = y + dy - 7;
        if (ky < 0 || ky >= HH) continue;          // uniform over wave
        #pragma unroll 1
        for (int half = 0; half < 2; ++half) {
            const int kx = x0 - 7 + half * 16 + l15;
            const bool cv = (kx >= 0) && (kx < WW);
            const float* kp = k + ((size_t)((n * NH + g) * DA + kbB)) * HW + ky * WW + kx;
            v16bf b0, b1;
            #pragma unroll
            for (int e = 0; e < 16; ++e) {
                float v0 = cv ? kp[(size_t)e * HW] : 0.f;
                float v1 = cv ? kp[(size_t)(e + 32) * HW] : 0.f;
                b0[e] = u2bf(f2bfu(v0));
                b1[e] = u2bf(f2bfu(v1));
            }
            v8f acc = {};
            acc = wmma_bf(a0, b0, acc);
            acc = wmma_bf(a1, b1, acc);
            const int t  = l15;
            const int mh = (lane >> 4) * 8;
            #pragma unroll
            for (int r = 0; r < 8; ++r) {
                const int p  = r + mh;
                const int dxs = t + half * 16 - p;   // dx + 7
                if (dxs >= 0 && dxs <= 14)
                    sc[p * WSP + dy * 16 + dxs] += acc[r];
            }
        }
    }
    __syncthreads();

    // ---- masked softmax over 240 slots (2 lanes per position) ----
    {
        const int p = l15, h = lane >> 4;
        const int x = x0 + p;
        const int pos = y * WW + x;
        const float* brg = b_rel + g * WSQ;

        float mx = -1e30f;
        #pragma unroll 1
        for (int j = h; j < WSP; j += 2) {
            const int dy = j >> 4, dxs = j & 15;
            const bool valid = (dxs < 15) &&
                               ((unsigned)(y + dy - 7) < (unsigned)HH) &&
                               ((unsigned)(x + dxs - 7) < (unsigned)WW);
            if (valid) mx = fmaxf(mx, sc[p * WSP + j] + brg[dy * 15 + dxs]);
        }
        mx = fmaxf(mx, __shfl_xor(mx, 16, 32));

        float s = 0.f;
        #pragma unroll 1
        for (int j = h; j < WSP; j += 2) {
            const int dy = j >> 4, dxs = j & 15;
            const bool valid = (dxs < 15) &&
                               ((unsigned)(y + dy - 7) < (unsigned)HH) &&
                               ((unsigned)(x + dxs - 7) < (unsigned)WW);
            float e = 0.f;
            if (valid) e = __expf(sc[p * WSP + j] + brg[dy * 15 + dxs] - mx);
            sc[p * WSP + j] = e;
            s += e;
        }
        s += __shfl_xor(s, 16, 32);
        const float inv = 1.f / s;

        unsigned short* ab = attn + ((size_t)((n * NH + g) * HW + pos)) * WSP;
        #pragma unroll 1
        for (int j = h; j < WSP; j += 2) ab[j] = f2bfu(sc[p * WSP + j] * inv);
    }
}

// ---------------------------------------------------------------------------
// Kernel 2: aggregation  agg[c, x] = sum_dy sum_t v[c, ky, x0-7+t] * attnBand[t, x]
// one wave per (n, g, y, x-tile); 15 banded B tiles preloaded, 16 c-tiles
// ---------------------------------------------------------------------------
__global__ __launch_bounds__(32) void k_agg(const unsigned short* __restrict__ attn,
                                            const float* __restrict__ v,
                                            float* __restrict__ agg)
{
    int b = blockIdx.x;
    const int xt = b % 6;  b /= 6;
    const int y  = b % 96; b /= 96;
    const int g  = b % 2;
    const int n  = b / 2;

    const int lane = threadIdx.x;
    const int l15  = lane & 15;
    const int kbA  = (lane >> 4) << 3;
    const int kbB  = (lane >> 4) << 4;
    const int x0   = xt * 16;

    // B tiles: Bt[dy][x=l15, t] = attn[pos(x)][dy*16 + (t - l15)]  (band; slot15==0)
    v16bf Bt[15];
    {
        const int pos = y * WW + x0 + l15;
        const unsigned short* ab = attn + ((size_t)((n * NH + g) * HW + pos)) * WSP;
        #pragma unroll
        for (int dy = 0; dy < 15; ++dy) {
            #pragma unroll
            for (int e = 0; e < 16; ++e) {
                const int t   = kbB + e;
                const int idx = t - l15;
                Bt[dy][e] = (idx >= 0 && idx < 16) ? u2bf(ab[dy * 16 + idx])
                                                   : u2bf((unsigned short)0);
            }
        }
    }

    #pragma unroll 1
    for (int ct = 0; ct < 16; ++ct) {
        v8f acc = {};
        const int c = g * HID_ + ct * 16 + l15;
        #pragma unroll
        for (int dy = 0; dy < 15; ++dy) {
            const int ky = y + dy - 7;
            const bool rv = (ky >= 0) && (ky < HH);
            const float* vb = v + ((size_t)(n * DV + c)) * HW + ky * WW;
            v16bf a;
            #pragma unroll
            for (int e = 0; e < 16; ++e) {
                const int t  = kbB + e;
                const int kx = x0 - 7 + t;
                const float vv = (rv && kx >= 0 && kx < WW) ? vb[kx] : 0.f;
                a[e] = u2bf(f2bfu(vv));
            }
            acc = wmma_bf(a, Bt[dy], acc);
        }
        const int mh = (lane >> 4) * 8;
        float* ob = agg + ((size_t)(n * DV + g * HID_ + ct * 16)) * HW + y * WW + x0 + l15;
        #pragma unroll
        for (int r = 0; r < 8; ++r) ob[(size_t)(r + mh) * HW] = acc[r];
    }
}

// ---------------------------------------------------------------------------
// Kernel 3: depthwise 5x5 conv (pad 2) on agg[n,512,96,96]; write transposed
//           bf16 xct[n][pos][512] (GEMM-friendly A layout for projection)
// ---------------------------------------------------------------------------
__global__ __launch_bounds__(256) void k_dwconv(const float* __restrict__ agg,
                                                const float* __restrict__ w_dw,
                                                unsigned short* __restrict__ xct)
{
    const int t = blockIdx.x * 256 + threadIdx.x;
    const int pos = t % HW;
    const int ch  = (t / HW) % DV;
    const int n   = t / (HW * DV);
    const int x = pos % WW, y = pos / WW;
    const float* ab = agg + ((size_t)(n * DV + ch)) * HW;
    const float* wd = w_dw + ch * 25;
    float s = 0.f;
    #pragma unroll
    for (int i = 0; i < 5; ++i) {
        const int yy = y + i - 2;
        if (yy < 0 || yy >= HH) continue;
        #pragma unroll
        for (int j = 0; j < 5; ++j) {
            const int xx = x + j - 2;
            if (xx < 0 || xx >= WW) continue;
            s += ab[yy * WW + xx] * wd[i * 5 + j];
        }
    }
    xct[((size_t)(n * HW + pos)) * DV + ch] = f2bfu(s);
}

// ---------------------------------------------------------------------------
// Kernel 4: projection  out[pos, n, o] = sum_i x[pos,i] * w_proj[o,i] + b_proj[o]
// one wave per (n, pos-tile of 16, 64 outputs); K=512 chained WMMA
// ---------------------------------------------------------------------------
__global__ __launch_bounds__(32) void k_proj(const unsigned short* __restrict__ xct,
                                             const float* __restrict__ w_proj,
                                             const float* __restrict__ b_proj,
                                             float* __restrict__ out)
{
    int b = blockIdx.x;
    const int ob = (b % 8) * 64; b /= 8;
    const int pt = b % 576;
    const int n  = b / 576;

    const int lane = threadIdx.x;
    const int l15  = lane & 15;
    const int kbA  = (lane >> 4) << 3;
    const int kbB  = (lane >> 4) << 4;
    const int pos0 = pt * 16;

    v8f acc[4] = {};
    #pragma unroll 1
    for (int kk = 0; kk < 16; ++kk) {
        v16bf a;
        const unsigned short* xb =
            xct + ((size_t)(n * HW + pos0 + l15)) * DV + kk * 32 + kbB;
        #pragma unroll
        for (int e = 0; e < 16; ++e) a[e] = u2bf(xb[e]);
        #pragma unroll
        for (int og = 0; og < 4; ++og) {
            const int o = ob + og * 16 + l15;
            const float* wb = w_proj + (size_t)o * DV + kk * 32 + kbB;
            v16bf bf_;
            #pragma unroll
            for (int e = 0; e < 16; ++e) bf_[e] = u2bf(f2bfu(wb[e]));
            acc[og] = wmma_bf(a, bf_, acc[og]);
        }
    }
    const int mh = (lane >> 4) * 8;
    #pragma unroll
    for (int og = 0; og < 4; ++og) {
        const int o = ob + og * 16 + l15;
        const float bp = b_proj[o];
        #pragma unroll
        for (int r = 0; r < 8; ++r) {
            const int pos = pos0 + r + mh;
            out[((size_t)pos * N_ + n) * DV + o] = acc[og][r] + bp;
        }
    }
}

// ---------------------------------------------------------------------------
extern "C" void kernel_launch(void* const* d_in, const int* in_sizes, int n_in,
                              void* d_out, int out_size, void* d_ws, size_t ws_size,
                              hipStream_t stream)
{
    (void)in_sizes; (void)n_in; (void)out_size; (void)ws_size;
    const float* q      = (const float*)d_in[0];
    const float* k      = (const float*)d_in[1];
    const float* v      = (const float*)d_in[2];
    const float* w_rel  = (const float*)d_in[3];
    const float* b_rel  = (const float*)d_in[4];
    const float* w_dw   = (const float*)d_in[5];
    const float* w_proj = (const float*)d_in[6];
    const float* b_proj = (const float*)d_in[7];
    float* out = (float*)d_out;

    char* ws = (char*)d_ws;
    const size_t attn_bytes = (size_t)N_ * NH * HW * WSP * 2;   // 35,389,440
    const size_t agg_bytes  = (size_t)N_ * DV * HW * 4;         // 75,497,472
    unsigned short* attn = (unsigned short*)ws;
    float*          agg  = (float*)(ws + attn_bytes);
    unsigned short* xct  = (unsigned short*)(ws + attn_bytes + agg_bytes);

    const int nb_tiles = N_ * NH * HH * (WW / 16);              // 4608
    k_scores<<<nb_tiles, 32, 0, stream>>>(q, k, w_rel, b_rel, attn);
    k_agg   <<<nb_tiles, 32, 0, stream>>>(attn, v, agg);
    k_dwconv<<<(N_ * DV * HW) / 256, 256, 0, stream>>>(agg, w_dw, xct);
    k_proj  <<<N_ * (HW / 16) * (DV / 64), 32, 0, stream>>>(xct, w_proj, b_proj, out);
}
